// CrossAttention_28338194219738
// MI455X (gfx1250) — compile-verified, matrix operand order checked by reading, not run on hardware
//
#include <hip/hip_runtime.h>
#include <hip/hip_bf16.h>
#include <stdint.h>

// Problem dims (fixed by reference)
#define BB   8
#define NN   1024
#define SS   1024
#define DIMV 1024
#define HH   16
#define DHH  64
#define HD   1024          // H*DH
#define MTOT (BB*NN)       // 8192 rows for all GEMMs

typedef __attribute__((ext_vector_type(16))) __bf16 v16bf;
typedef __attribute__((ext_vector_type(8)))  float  v8f;
typedef __attribute__((ext_vector_type(4)))  __bf16 bf16x4;

__device__ __forceinline__ v8f zero8() {
  v8f z;
#pragma unroll
  for (int e = 0; e < 8; ++e) z[e] = 0.0f;
  return z;
}

// Load one 16x32 bf16 WMMA A/B fragment from a row-major, K-contiguous source
// (global or LDS). Layout per ISA 16-bit A-matrix table:
//   lanes 0-15:  K = k0+{0..7}   and K = k0+16+{0..7}
//   lanes 16-31: K = k0+8+{0..7} and K = k0+24+{0..7}
__device__ __forceinline__ v16bf load_frag(const __bf16* rowp, int k0, int lhi) {
  union { v16bf v; float4 f[2]; } u;
  const float4* q = (const float4*)(rowp + k0 + lhi * 8);
  u.f[0] = q[0];
  u.f[1] = q[2];   // +32 bytes
  return u.v;
}

// Async copy 16 bytes global -> LDS (per active lane). Tracked by ASYNCcnt.
__device__ __forceinline__ void async_b128(uint32_t lds_addr, uint32_t voff,
                                           const __bf16* sbase) {
  asm volatile("global_load_async_to_lds_b128 %0, %1, %2"
               :: "v"(lds_addr), "v"(voff), "s"(sbase)
               : "memory");
}

__device__ __forceinline__ uint32_t lds_u32(const void* p) {
  return (uint32_t)(uint64_t)(uintptr_t)p;
}

// ---------------------------------------------------------------------------
// fp32 -> bf16 cast (vectorized x4)
// ---------------------------------------------------------------------------
__global__ __launch_bounds__(256) void cast_f32_bf16(const float* __restrict__ in,
                                                     __bf16* __restrict__ out, int n4) {
  int i = blockIdx.x * blockDim.x + threadIdx.x;
  if (i < n4) {
    float4 f = ((const float4*)in)[i];
    bf16x4 o;
    o[0] = (__bf16)f.x; o[1] = (__bf16)f.y; o[2] = (__bf16)f.z; o[3] = (__bf16)f.w;
    ((bf16x4*)out)[i] = o;
  }
}

// ---------------------------------------------------------------------------
// GEMM: C[8192,1024] = A[8192,1024] (bf16) x Bm[1024,1024]^T (bf16, = W rows)
// Double-buffered async-to-LDS staging of 128x32 A/B k-slices, fragments via
// ds_load_b128, fp32 accumulate with v_wmma_f32_16x16x32_bf16.
// Block = 128x128 tile, 8 waves (2 M x 4 N), wave = 64x32 = 4x2 WMMA tiles.
// ---------------------------------------------------------------------------
template <bool F32OUT>
__global__ __launch_bounds__(256) void gemm_wmma_bf16(
    const __bf16* __restrict__ A, const __bf16* __restrict__ Bm,
    float* __restrict__ Cf, __bf16* __restrict__ Cb,
    const float* __restrict__ bias) {
  __shared__ __bf16 Al[2][128][32];   // 8 KB per buffer
  __shared__ __bf16 Bl[2][128][32];   // 8 KB per buffer

  const int tid  = threadIdx.x;
  const int lane = tid & 31;
  const int wave = tid >> 5;
  const int wm = wave >> 2, wn = wave & 3;
  const int lrow = lane & 15, lhi = lane >> 4;

  // async staging: thread t covers tile row t>>1, byte segment (t&1)*32 of 64B row
  const int crow = tid >> 1;
  const int cseg = (tid & 1) * 32;                       // bytes within 32-elem row
  const __bf16* Abase = A  + (size_t)(blockIdx.x * 128) * DIMV;
  const __bf16* Bbase = Bm + (size_t)(blockIdx.y * 128) * DIMV;
  const uint32_t voff = (uint32_t)(crow * (DIMV * 2) + cseg);

  auto issue = [&](int buf, int k) {
    const __bf16* sa = Abase + k;
    const __bf16* sb = Bbase + k;
    const uint32_t la = lds_u32(&Al[buf][crow][0]) + cseg;
    const uint32_t lb = lds_u32(&Bl[buf][crow][0]) + cseg;
    async_b128(la,      voff,      sa);
    async_b128(la + 16, voff + 16, sa);
    async_b128(lb,      voff,      sb);
    async_b128(lb + 16, voff + 16, sb);
  };

  v8f acc[4][2];
#pragma unroll
  for (int i = 0; i < 4; ++i)
#pragma unroll
    for (int j = 0; j < 2; ++j) acc[i][j] = zero8();

  issue(0, 0);
  const int steps = DIMV / 32;   // 32
  for (int s = 0; s < steps; ++s) {
    const int buf = s & 1;
    if (s + 1 < steps) {
      issue(buf ^ 1, (s + 1) * 32);
      asm volatile("s_wait_asynccnt 0x4" ::: "memory");  // current buffer done
    } else {
      asm volatile("s_wait_asynccnt 0x0" ::: "memory");
    }
    __syncthreads();

    v16bf a[4], b[2];
#pragma unroll
    for (int i = 0; i < 4; ++i)
      a[i] = load_frag(&Al[buf][wm * 64 + i * 16 + lrow][0], 0, lhi);
#pragma unroll
    for (int j = 0; j < 2; ++j)
      b[j] = load_frag(&Bl[buf][wn * 32 + j * 16 + lrow][0], 0, lhi);
#pragma unroll
    for (int i = 0; i < 4; ++i)
#pragma unroll
      for (int j = 0; j < 2; ++j)
        acc[i][j] = __builtin_amdgcn_wmma_f32_16x16x32_bf16(
            false, a[i], false, b[j], (short)0, acc[i][j], false, false);

    __syncthreads();   // readers done before this buffer is refilled
  }

  // C layout: VGPR r of lane L -> (m = r + 8*(L>>4), n = L&15) within 16x16 tile
  const int m0 = blockIdx.x * 128 + wm * 64;
  const int n0 = blockIdx.y * 128 + wn * 32;
#pragma unroll
  for (int i = 0; i < 4; ++i) {
    const int mb = m0 + i * 16 + 8 * lhi;
#pragma unroll
    for (int j = 0; j < 2; ++j) {
      const int nidx = n0 + j * 16 + lrow;
      if (F32OUT) {
        const float bv = bias[nidx];
#pragma unroll
        for (int r = 0; r < 8; ++r)
          Cf[(size_t)(mb + r) * DIMV + nidx] = acc[i][j][r] + bv;
      } else {
#pragma unroll
        for (int r = 0; r < 8; ++r)
          Cb[(size_t)(mb + r) * DIMV + nidx] = (__bf16)acc[i][j][r];
      }
    }
  }
}

// ---------------------------------------------------------------------------
// Flash attention over the reshaped [B*H, N|S, DH=64] views.
// Block = 128 Q rows of one (b,h); 8 waves x 16 rows. S chunks of 64.
// Double-buffered staging: K via async-to-LDS, V transposed via registers+LDS,
// both overlapped with the previous chunk's WMMAs/softmax.
// Output written directly in [B, N, H*DH] (post-transpose) layout as bf16.
// ---------------------------------------------------------------------------
__global__ __launch_bounds__(256) void attn_wmma(
    const __bf16* __restrict__ Qp, const __bf16* __restrict__ Kp,
    const __bf16* __restrict__ Vp, __bf16* __restrict__ Om) {
  __shared__ __bf16 Kl[2][64][64];    // K chunk, row-major [s][d]   (2 x 8 KB)
  __shared__ __bf16 Vt[2][DHH][72];   // V chunk transposed [d][s], padded
  __shared__ __bf16 Pl[8][16][72];    // per-wave P staging (16 x 64, padded)

  const int tid  = threadIdx.x;
  const int lane = tid & 31;
  const int wave = tid >> 5;
  const int lrow = lane & 15, lhi = lane >> 4;
  const int b = blockIdx.z, h = blockIdx.y;
  const int qrow0 = blockIdx.x * 128 + wave * 16;
  const size_t bh = (size_t)(b * HH + h);
  const __bf16* Qb = Qp + bh * NN * DHH;
  const __bf16* Kb = Kp + bh * SS * DHH;
  const __bf16* Vb = Vp + bh * SS * DHH;

  // async K staging: thread t covers s-row t>>2, byte segment (t&3)*32 of 128B row
  const int krow = tid >> 2;
  const int kseg = (tid & 3) * 32;
  const uint32_t kvoff = (uint32_t)(krow * (DHH * 2) + kseg);
  auto issueK = [&](int buf, int sc) {
    const __bf16* ksrc = Kb + (size_t)sc * DHH;
    const uint32_t kl = lds_u32(&Kl[buf][krow][0]) + kseg;
    async_b128(kl,      kvoff,      ksrc);
    async_b128(kl + 16, kvoff + 16, ksrc);
  };

  // V staging through registers: thread t covers s-row t>>2, d-quarter (t&3)*16
  const int sl = tid >> 2;
  const int dq = (tid & 3) * 16;
  union VU { float4 f[2]; __bf16 hv[16]; };
  auto loadV = [&](VU& u, int sc) {
    const float4* src = (const float4*)(Vb + (size_t)(sc + sl) * DHH + dq);
    u.f[0] = src[0]; u.f[1] = src[1];
  };
  auto storeV = [&](int buf, const VU& u) {
#pragma unroll
    for (int e = 0; e < 16; ++e) Vt[buf][dq + e][sl] = u.hv[e];
  };

  // Q fragments (DH=64 -> 2 k-steps), loaded once
  v16bf qf[2];
  qf[0] = load_frag(Qb + (size_t)(qrow0 + lrow) * DHH, 0, lhi);
  qf[1] = load_frag(Qb + (size_t)(qrow0 + lrow) * DHH, 32, lhi);

  v8f o[4];
#pragma unroll
  for (int dt = 0; dt < 4; ++dt) o[dt] = zero8();
  float mrun[8], lrun[8];
#pragma unroll
  for (int r = 0; r < 8; ++r) { mrun[r] = -3.0e38f; lrun[r] = 0.0f; }

  // prologue: start chunk 0 staging
  VU vu;
  issueK(0, 0);
  loadV(vu, 0);

  const int chunks = SS / 64;   // 16
  for (int c = 0; c < chunks; ++c) {
    const int buf = c & 1;

    // commit V registers -> LDS; start next chunk's staging (overlaps compute wait)
    storeV(buf, vu);
    if (c + 1 < chunks) {
      issueK(buf ^ 1, (c + 1) * 64);
      loadV(vu, (c + 1) * 64);
      asm volatile("s_wait_asynccnt 0x2" ::: "memory");  // current K buffer done
    } else {
      asm volatile("s_wait_asynccnt 0x0" ::: "memory");
    }
    __syncthreads();

    // scores S = Q K^T * scale  (16 x 64 per wave, 4 s-tiles), K frags from LDS
    v8f sa[4];
#pragma unroll
    for (int st = 0; st < 4; ++st) {
      sa[st] = zero8();
      const __bf16* krowp = &Kl[buf][st * 16 + lrow][0];
      v16bf b0 = load_frag(krowp, 0, lhi);
      v16bf b1 = load_frag(krowp, 32, lhi);
      sa[st] = __builtin_amdgcn_wmma_f32_16x16x32_bf16(false, qf[0], false, b0,
                                                       (short)0, sa[st], false, false);
      sa[st] = __builtin_amdgcn_wmma_f32_16x16x32_bf16(false, qf[1], false, b1,
                                                       (short)0, sa[st], false, false);
    }

    // chunk row-max (rows live in lane groups of 16; xor-reduce over n)
    float cm[8];
#pragma unroll
    for (int r = 0; r < 8; ++r) cm[r] = -3.0e38f;
#pragma unroll
    for (int st = 0; st < 4; ++st)
#pragma unroll
      for (int r = 0; r < 8; ++r) {
        sa[st][r] *= 0.125f;  // DH^-0.5
        cm[r] = fmaxf(cm[r], sa[st][r]);
      }
#pragma unroll
    for (int mask = 1; mask <= 8; mask <<= 1)
#pragma unroll
      for (int r = 0; r < 8; ++r) cm[r] = fmaxf(cm[r], __shfl_xor(cm[r], mask, 32));

    // online softmax update
    float mnew[8], scf[8], rs[8];
#pragma unroll
    for (int r = 0; r < 8; ++r) {
      mnew[r] = fmaxf(mrun[r], cm[r]);
      scf[r] = __expf(mrun[r] - mnew[r]);
      rs[r] = 0.0f;
    }
#pragma unroll
    for (int st = 0; st < 4; ++st)
#pragma unroll
      for (int r = 0; r < 8; ++r) {
        sa[st][r] = __expf(sa[st][r] - mnew[r]);  // sa becomes P
        rs[r] += sa[st][r];
      }
#pragma unroll
    for (int mask = 1; mask <= 8; mask <<= 1)
#pragma unroll
      for (int r = 0; r < 8; ++r) rs[r] += __shfl_xor(rs[r], mask, 32);
#pragma unroll
    for (int r = 0; r < 8; ++r) {
      lrun[r] = lrun[r] * scf[r] + rs[r];
      mrun[r] = mnew[r];
    }
#pragma unroll
    for (int dt = 0; dt < 4; ++dt)
#pragma unroll
      for (int r = 0; r < 8; ++r) o[dt][r] *= scf[r];

    // P (C-layout) -> LDS bf16 (row-major) for A-frag reuse (same-wave region)
#pragma unroll
    for (int st = 0; st < 4; ++st)
#pragma unroll
      for (int r = 0; r < 8; ++r)
        Pl[wave][8 * lhi + r][st * 16 + lrow] = (__bf16)sa[st][r];

    // O += P @ V  (A = P from LDS, B = Vt rows = contiguous s)
#pragma unroll
    for (int kk = 0; kk < 2; ++kk) {
      v16bf pa = load_frag(&Pl[wave][lrow][0], kk * 32, lhi);
#pragma unroll
      for (int dt = 0; dt < 4; ++dt) {
        v16bf vb = load_frag(&Vt[buf][dt * 16 + lrow][0], kk * 32, lhi);
        o[dt] = __builtin_amdgcn_wmma_f32_16x16x32_bf16(false, pa, false, vb,
                                                        (short)0, o[dt], false, false);
      }
    }

    __syncthreads();  // all waves done with Kl[buf]/Vt[buf] before refill at c+2
  }

  // normalize and scatter into [B, N, H*DH] (transpose(0,2,1,3) fused here)
  float inv[8];
#pragma unroll
  for (int r = 0; r < 8; ++r) inv[r] = 1.0f / lrun[r];
#pragma unroll
  for (int dt = 0; dt < 4; ++dt) {
    const int col = h * DHH + dt * 16 + lrow;
#pragma unroll
    for (int r = 0; r < 8; ++r) {
      const int n2 = qrow0 + 8 * lhi + r;
      Om[((size_t)b * NN + n2) * HD + col] = (__bf16)(o[dt][r] * inv[r]);
    }
  }
}

// ---------------------------------------------------------------------------
// Launch
// ---------------------------------------------------------------------------
extern "C" void kernel_launch(void* const* d_in, const int* in_sizes, int n_in,
                              void* d_out, int out_size, void* d_ws, size_t ws_size,
                              hipStream_t stream) {
  const float* q  = (const float*)d_in[0];
  const float* k  = (const float*)d_in[1];
  const float* v  = (const float*)d_in[2];
  const float* Wq = (const float*)d_in[3];
  const float* Wk = (const float*)d_in[4];
  const float* Wv = (const float*)d_in[5];
  const float* Wo = (const float*)d_in[6];
  const float* bo = (const float*)d_in[7];

  char* ws = (char*)d_ws;
  size_t off = 0;
  auto alloc = [&](size_t bytes) {
    char* p = ws + off;
    off += (bytes + 255) & ~(size_t)255;
    return p;
  };
  const size_t actBytes = (size_t)MTOT * DIMV * sizeof(__bf16);  // 16 MB
  const size_t wBytes   = (size_t)HD * DIMV * sizeof(__bf16);    //  2 MB
  __bf16* q_bf  = (__bf16*)alloc(actBytes);
  __bf16* k_bf  = (__bf16*)alloc(actBytes);
  __bf16* v_bf  = (__bf16*)alloc(actBytes);
  __bf16* wq_bf = (__bf16*)alloc(wBytes);
  __bf16* wk_bf = (__bf16*)alloc(wBytes);
  __bf16* wv_bf = (__bf16*)alloc(wBytes);
  __bf16* wo_bf = (__bf16*)alloc(wBytes);
  __bf16* Qp    = (__bf16*)alloc(actBytes);
  __bf16* Kp    = (__bf16*)alloc(actBytes);
  __bf16* Vp    = (__bf16*)alloc(actBytes);
  __bf16* Omid  = (__bf16*)alloc(actBytes);

  // casts
  const int nAct4 = (MTOT * DIMV) / 4;
  const int nW4   = (HD * DIMV) / 4;
  cast_f32_bf16<<<nAct4 / 256, 256, 0, stream>>>(q, q_bf, nAct4);
  cast_f32_bf16<<<nAct4 / 256, 256, 0, stream>>>(k, k_bf, nAct4);
  cast_f32_bf16<<<nAct4 / 256, 256, 0, stream>>>(v, v_bf, nAct4);
  cast_f32_bf16<<<nW4 / 256, 256, 0, stream>>>(Wq, wq_bf, nW4);
  cast_f32_bf16<<<nW4 / 256, 256, 0, stream>>>(Wk, wk_bf, nW4);
  cast_f32_bf16<<<nW4 / 256, 256, 0, stream>>>(Wv, wv_bf, nW4);
  cast_f32_bf16<<<nW4 / 256, 256, 0, stream>>>(Wo, wo_bf, nW4);

  // projections: [8192,1024] x [1024,1024]^T -> bf16
  dim3 gg(MTOT / 128, DIMV / 128);  // 64 x 8
  gemm_wmma_bf16<false><<<gg, 256, 0, stream>>>(q_bf, wq_bf, nullptr, Qp, nullptr);
  gemm_wmma_bf16<false><<<gg, 256, 0, stream>>>(k_bf, wk_bf, nullptr, Kp, nullptr);
  gemm_wmma_bf16<false><<<gg, 256, 0, stream>>>(v_bf, wv_bf, nullptr, Vp, nullptr);

  // attention (flash), output fused into [B, N, H*DH] layout
  attn_wmma<<<dim3(NN / 128, HH, BB), 256, 0, stream>>>(Qp, Kp, Vp, Omid);

  // output projection: f32 + bias -> d_out
  gemm_wmma_bf16<true><<<gg, 256, 0, stream>>>(Omid, wo_bf, (float*)d_out, nullptr, bo);
}
